// MultiHeadSelfAttention_47261820125675
// MI455X (gfx1250) — compile-verified, matrix operand order checked by reading, not run on hardware
//
#include <hip/hip_runtime.h>

#define DM    1024
#define NHEAD 16
#define DH    64
#define NB    2
#define TSEQ  2048
#define MTOT  (NB*TSEQ)      // 4096
#define N_QKV (3*DM)         // 3072
#define KTILE 32
#define MW    (TSEQ/32)      // 64 mask words per row
#define NKT   (TSEQ/32)      // 64 key tiles in attention

typedef __attribute__((ext_vector_type(16))) __bf16 v16bf;
typedef __attribute__((ext_vector_type(8)))  __bf16 v8bf;
typedef __attribute__((ext_vector_type(8)))  float  v8f;

union Frag16 { v16bf v; v8bf h[2]; };

__device__ __forceinline__ v8f wmma_bf16(v16bf a, v16bf b, v8f c) {
  // D = A(16x32) * B(32x16) + C, f32 accumulate
  return __builtin_amdgcn_wmma_f32_16x16x32_bf16(false, a, false, b, (short)0, c,
                                                 false, false);
}

__device__ __forceinline__ float redmax16(float v) {
#pragma unroll
  for (int m = 1; m < 16; m <<= 1) v = fmaxf(v, __shfl_xor(v, m, 16));
  return v;
}
__device__ __forceinline__ float redsum16(float v) {
#pragma unroll
  for (int m = 1; m < 16; m <<= 1) v += __shfl_xor(v, m, 16);
  return v;
}

// ---------------- f32 -> bf16 flat convert (x) ----------------
__global__ void cvt_bf16_kernel(const float* __restrict__ s,
                                __bf16* __restrict__ d, int n4) {
  int i = blockIdx.x * blockDim.x + threadIdx.x;
  if (i >= n4) return;
  float4 v = ((const float4*)s)[i];
  __bf16* p = d + (size_t)i * 4;
  p[0] = (__bf16)v.x; p[1] = (__bf16)v.y; p[2] = (__bf16)v.z; p[3] = (__bf16)v.w;
}

// ---------------- weight swizzle: [K=1024, N] f32 -> B-fragment order bf16 ----
// out[(kt*NT + nt)*512 + lane*16 + j], j follows v16bf element order:
//   k = kt*32 + 16*(v>>2) + 8*(lane>>4) + 2*(v&3) + (j&1), v = j>>1
__global__ void swz_w_kernel(const float* __restrict__ W,
                             __bf16* __restrict__ out, int N) {
  int tid = blockIdx.x * blockDim.x + threadIdx.x;
  int NT = N >> 4;
  int total = (DM / KTILE) * NT * 32;
  if (tid >= total) return;
  int lane = tid & 31;
  int rest = tid >> 5;
  int nt = rest % NT;
  int kt = rest / NT;
  int n = nt * 16 + (lane & 15);
  int half = lane >> 4;
  __align__(16) __bf16 tmp[16];
#pragma unroll
  for (int j = 0; j < 16; ++j) {
    int v = j >> 1;
    int k = kt * KTILE + 16 * (v >> 2) + 8 * half + 2 * (v & 3) + (j & 1);
    tmp[j] = (__bf16)W[(size_t)k * N + n];
  }
  __bf16* dst = out + ((size_t)(kt * NT + nt) * 512 + lane * 16);
  *(v8bf*)dst       = *(const v8bf*)tmp;
  *(v8bf*)(dst + 8) = *(const v8bf*)(tmp + 8);
}

// ---------------- mask [T,T] int32 -> bitmask words --------------------------
__global__ void pack_mask_kernel(const int* __restrict__ mask,
                                 unsigned* __restrict__ bits) {
  int tid = blockIdx.x * blockDim.x + threadIdx.x;
  if (tid >= TSEQ * MW) return;
  int t = tid >> 6, w = tid & 63;
  const int* p = mask + (size_t)t * TSEQ + w * 32;
  unsigned acc = 0;
#pragma unroll 8
  for (int i = 0; i < 32; ++i) acc |= (unsigned)(p[i] == 1) << i;
  bits[tid] = acc;
}

// -------- GEMM building blocks: 32M x 64N per wave, double-buffered ----------
__device__ __forceinline__ void load_tile2(const __bf16* __restrict__ arow0,
                                           const __bf16* __restrict__ arow1,
                                           const __bf16* __restrict__ Wswz,
                                           int kt, int NT, int ntb, int lane,
                                           Frag16* aa, Frag16* bb) {
  aa[0].h[0] = *(const v8bf*)(arow0 + kt * KTILE);
  aa[0].h[1] = *(const v8bf*)(arow0 + kt * KTILE + 16);
  aa[1].h[0] = *(const v8bf*)(arow1 + kt * KTILE);
  aa[1].h[1] = *(const v8bf*)(arow1 + kt * KTILE + 16);
  const __bf16* bbase = Wswz + ((size_t)kt * NT + ntb) * 512 + lane * 16;
#pragma unroll
  for (int nt = 0; nt < 4; ++nt) {
    const __bf16* bp = bbase + (size_t)nt * 512;
    bb[nt].h[0] = *(const v8bf*)bp;
    bb[nt].h[1] = *(const v8bf*)(bp + 8);
  }
}
__device__ __forceinline__ void mma2x4(const Frag16* aa, const Frag16* bb,
                                       v8f* acc) {  // acc[mt*4+nt]
#pragma unroll
  for (int nt = 0; nt < 4; ++nt) {
    acc[nt]     = wmma_bf16(aa[0].v, bb[nt].v, acc[nt]);
    acc[4 + nt] = wmma_bf16(aa[1].v, bb[nt].v, acc[4 + nt]);
  }
}

// ---------------- qkv GEMM: Xb[4096,1024] @ Wswz -> Q(scaled)/K/V^T bf16 ------
__global__ __launch_bounds__(256) void qkv_gemm_kernel(
    const __bf16* __restrict__ Xb, const __bf16* __restrict__ Wswz,
    const float* __restrict__ bias, __bf16* __restrict__ Qs,
    __bf16* __restrict__ Kmat, __bf16* __restrict__ VT) {
  const int lane = threadIdx.x & 31;
  const int wave = threadIdx.x >> 5;
  const int c16 = lane & 15;
  const int half = lane >> 4;
  const int wm = wave & 1, wn = wave >> 1;
  const int mb = blockIdx.y * 64 + wm * 32;
  const int ntb = blockIdx.x * 16 + wn * 4;  // in 16-col tile units
  const int NT = N_QKV / 16;                 // 192

  v8f acc[8] = {};
  Frag16 a[2][2], b[2][4];
  const __bf16* arow0 = Xb + (size_t)(mb + c16) * DM + 8 * half;
  const __bf16* arow1 = arow0 + (size_t)16 * DM;

  load_tile2(arow0, arow1, Wswz, 0, NT, ntb, lane, a[0], b[0]);
#pragma unroll 1
  for (int kt = 0; kt < DM / KTILE; kt += 2) {
    load_tile2(arow0, arow1, Wswz, kt + 1, NT, ntb, lane, a[1], b[1]);
    mma2x4(a[0], b[0], acc);
    if (kt + 2 < DM / KTILE)
      load_tile2(arow0, arow1, Wswz, kt + 2, NT, ntb, lane, a[0], b[0]);
    mma2x4(a[1], b[1], acc);
  }

  // epilogue: split into Q (x0.125), K, and transposed V
#pragma unroll
  for (int mt = 0; mt < 2; ++mt) {
    const int m0 = mb + mt * 16 + 8 * half;
    const int bidx = m0 >> 11;
    const int t0 = m0 & (TSEQ - 1);
#pragma unroll
    for (int nt = 0; nt < 4; ++nt) {
      const v8f& av = acc[mt * 4 + nt];
      const int col = (ntb + nt) * 16 + c16;
      const float bv = bias[col];
      const int part = col >> 10;
      const int hcol = col & (DM - 1);
      const int hh = hcol >> 6;
      const int dd = hcol & (DH - 1);
      const size_t bh = (size_t)bidx * NHEAD + hh;
      if (part == 0) {
#pragma unroll
        for (int r = 0; r < 8; ++r)
          Qs[(bh * TSEQ + t0 + r) * DH + dd] = (__bf16)((av[r] + bv) * 0.125f);
      } else if (part == 1) {
#pragma unroll
        for (int r = 0; r < 8; ++r)
          Kmat[(bh * TSEQ + t0 + r) * DH + dd] = (__bf16)(av[r] + bv);
      } else {
        __align__(16) __bf16 tmp[8];
#pragma unroll
        for (int r = 0; r < 8; ++r) tmp[r] = (__bf16)(av[r] + bv);
        *(v8bf*)&VT[(bh * DH + dd) * TSEQ + t0] = *(const v8bf*)tmp;
      }
    }
  }
}

// ---------------- block-cooperative flash attention --------------------------
// 8 waves share K/V tiles via double-buffered LDS; each wave owns 16 queries.
__global__ __launch_bounds__(256) void attn_kernel(
    const __bf16* __restrict__ Qs, const __bf16* __restrict__ Kmat,
    const __bf16* __restrict__ VT, const unsigned* __restrict__ mbits,
    __bf16* __restrict__ AO) {
  // padded rows: 144B (K) and 112B (V,P): 16B-aligned, 16 distinct bank-quads
  __shared__ __align__(16) __bf16 KtS[2][32][DH + 8];   // [buf][key][dh]
  __shared__ __align__(16) __bf16 VtS[2][DH][32 + 24];  // [buf][dh][key]
  __shared__ __align__(16) __bf16 Pst[8][16][32 + 24];  // per-wave P staging
  const int tid = threadIdx.x;
  const int lane = tid & 31, wave = tid >> 5;
  const int c16 = lane & 15, half = lane >> 4;
  const int bh = blockIdx.y;
  const int qb = blockIdx.x * 128 + wave * 16;
  const __bf16* Qp = Qs + (size_t)bh * TSEQ * DH;
  const __bf16* Kp = Kmat + (size_t)bh * TSEQ * DH;
  const __bf16* Vp = VT + (size_t)bh * DH * TSEQ;

  // per-thread staging coordinates (one 16B chunk of K and of V per thread)
  const int sk_key = tid >> 3, sk_ch = (tid & 7) * 8;  // K tile 32x64
  const int sv_d = tid >> 2, sv_ch = (tid & 3) * 8;    // V tile 64x32

  Frag16 qa[2];
#pragma unroll
  for (int s = 0; s < 2; ++s) {
    const __bf16* p = Qp + (size_t)(qb + c16) * DH + s * 32 + 8 * half;
    qa[s].h[0] = *(const v8bf*)p;
    qa[s].h[1] = *(const v8bf*)(p + 16);
  }

  v8f o[4] = {};
  float rm[8], rs[8];
#pragma unroll
  for (int r = 0; r < 8; ++r) { rm[r] = -1e30f; rs[r] = 0.f; }
  const int q0 = qb + 8 * half;

  // prologue: stage tile 0 into buffer 0
  {
    v8bf k0r = *(const v8bf*)(Kp + (size_t)sk_key * DH + sk_ch);
    v8bf v0r = *(const v8bf*)(Vp + (size_t)sv_d * TSEQ + sv_ch);
    *(v8bf*)&KtS[0][sk_key][sk_ch] = k0r;
    *(v8bf*)&VtS[0][sv_d][sv_ch] = v0r;
  }

  v8bf knext, vnext;
  for (int j = 0; j < NKT; ++j) {
    const int buf = j & 1;
    const int k0 = j * 32;
    __syncthreads();  // B0: tile `buf` fully staged; prev reads of buf^1 done
    if (j + 1 < NKT) {  // issue next tile's global loads early (regs only)
      knext = *(const v8bf*)(Kp + (size_t)(k0 + 32 + sk_key) * DH + sk_ch);
      vnext = *(const v8bf*)(Vp + (size_t)sv_d * TSEQ + (k0 + 32) + sv_ch);
    }
    unsigned mw[8];  // mask words prefetched; latency hides under QK wmmas
#pragma unroll
    for (int r = 0; r < 8; ++r) mw[r] = mbits[(size_t)(q0 + r) * MW + j];
    // ---- S = Q K^T from LDS (keys k0..k0+31) ----
    v8f s0 = {}, s1 = {};
#pragma unroll
    for (int s = 0; s < 2; ++s) {
      Frag16 b0, b1;
      const __bf16* kp0 = &KtS[buf][c16][s * 32 + 8 * half];
      b0.h[0] = *(const v8bf*)kp0;
      b0.h[1] = *(const v8bf*)(kp0 + 16);
      const __bf16* kp1 = &KtS[buf][16 + c16][s * 32 + 8 * half];
      b1.h[0] = *(const v8bf*)kp1;
      b1.h[1] = *(const v8bf*)(kp1 + 16);
      s0 = wmma_bf16(qa[s].v, b0.v, s0);
      s1 = wmma_bf16(qa[s].v, b1.v, s1);
    }
    // ---- masked online softmax ----
    float alpha[8];
#pragma unroll
    for (int r = 0; r < 8; ++r) {
      const unsigned w = mw[r];
      float v0 = ((w >> c16) & 1u) ? -1e30f : s0[r];
      float v1 = ((w >> (16 + c16)) & 1u) ? -1e30f : s1[r];
      float mx = redmax16(fmaxf(v0, v1));
      const float mnew = fmaxf(rm[r], mx);
      const float al = __expf(rm[r] - mnew);
      const float p0 = __expf(v0 - mnew);
      const float p1 = __expf(v1 - mnew);
      rm[r] = mnew;
      rs[r] = rs[r] * al + redsum16(p0 + p1);
      alpha[r] = al;
      Pst[wave][r + 8 * half][c16]      = (__bf16)p0;
      Pst[wave][r + 8 * half][16 + c16] = (__bf16)p1;
    }
#pragma unroll
    for (int nt = 0; nt < 4; ++nt)
#pragma unroll
      for (int r = 0; r < 8; ++r) o[nt][r] *= alpha[r];
    __syncthreads();  // B1: P writes visible (wave ds-ordering)
    Frag16 pa;        // P as A-fragment (rows=queries, K=32 keys)
    pa.h[0] = *(const v8bf*)&Pst[wave][c16][8 * half];
    pa.h[1] = *(const v8bf*)&Pst[wave][c16][16 + 8 * half];
#pragma unroll
    for (int nt = 0; nt < 4; ++nt) {
      Frag16 bv;  // V^T rows give contiguous B-fragments from LDS
      const __bf16* vp = &VtS[buf][nt * 16 + c16][8 * half];
      bv.h[0] = *(const v8bf*)vp;
      bv.h[1] = *(const v8bf*)(vp + 16);
      o[nt] = wmma_bf16(pa.v, bv.v, o[nt]);
    }
    // ---- late ds-stores of next tile: load latency hidden by whole iter ----
    if (j + 1 < NKT) {
      *(v8bf*)&KtS[buf ^ 1][sk_key][sk_ch] = knext;
      *(v8bf*)&VtS[buf ^ 1][sv_d][sv_ch] = vnext;
    }
  }
  const int b = bh >> 4, hh = bh & 15;
#pragma unroll
  for (int r = 0; r < 8; ++r) rs[r] = 1.f / fmaxf(rs[r], 1e-20f);
#pragma unroll
  for (int nt = 0; nt < 4; ++nt)
#pragma unroll
    for (int r = 0; r < 8; ++r) {
      const float v = o[nt][r] * rs[r];
      AO[(size_t)(b * TSEQ + q0 + r) * DM + hh * DH + nt * 16 + c16] = (__bf16)v;
    }
}

// ---------------- proj GEMM: AO[4096,1024] @ Wproj + bproj -> f32 out --------
__global__ __launch_bounds__(256) void proj_gemm_kernel(
    const __bf16* __restrict__ A, const __bf16* __restrict__ Wswz,
    const float* __restrict__ bias, float* __restrict__ out) {
  const int lane = threadIdx.x & 31;
  const int wave = threadIdx.x >> 5;
  const int c16 = lane & 15;
  const int half = lane >> 4;
  const int wm = wave & 1, wn = wave >> 1;
  const int mb = blockIdx.y * 64 + wm * 32;
  const int ntb = blockIdx.x * 16 + wn * 4;
  const int NT = DM / 16;  // 64

  v8f acc[8] = {};
  Frag16 a[2][2], b[2][4];
  const __bf16* arow0 = A + (size_t)(mb + c16) * DM + 8 * half;
  const __bf16* arow1 = arow0 + (size_t)16 * DM;

  load_tile2(arow0, arow1, Wswz, 0, NT, ntb, lane, a[0], b[0]);
#pragma unroll 1
  for (int kt = 0; kt < DM / KTILE; kt += 2) {
    load_tile2(arow0, arow1, Wswz, kt + 1, NT, ntb, lane, a[1], b[1]);
    mma2x4(a[0], b[0], acc);
    if (kt + 2 < DM / KTILE)
      load_tile2(arow0, arow1, Wswz, kt + 2, NT, ntb, lane, a[0], b[0]);
    mma2x4(a[1], b[1], acc);
  }

#pragma unroll
  for (int mt = 0; mt < 2; ++mt) {
    const int m0 = mb + mt * 16 + 8 * half;
#pragma unroll
    for (int nt = 0; nt < 4; ++nt) {
      const v8f& av = acc[mt * 4 + nt];
      const int col = (ntb + nt) * 16 + c16;
      const float bv = bias[col];
#pragma unroll
      for (int r = 0; r < 8; ++r)
        out[(size_t)(m0 + r) * DM + col] = av[r] + bv;
    }
  }
}

extern "C" void kernel_launch(void* const* d_in, const int* in_sizes, int n_in,
                              void* d_out, int out_size, void* d_ws, size_t ws_size,
                              hipStream_t stream) {
  const float* x     = (const float*)d_in[0];
  const int*   mask  = (const int*)d_in[1];
  const float* Wqkv  = (const float*)d_in[2];
  const float* bqkv  = (const float*)d_in[3];
  const float* Wproj = (const float*)d_in[4];
  const float* bproj = (const float*)d_in[5];
  float* out = (float*)d_out;

  char* w = (char*)d_ws;
  size_t off = 0;
  auto carve = [&](size_t bytes) {
    void* p = w + off;
    off += (bytes + 255) & ~(size_t)255;
    return p;
  };
  __bf16* Xb   = (__bf16*)carve((size_t)MTOT * DM * 2);        // 8 MB
  __bf16* Wq_s = (__bf16*)carve((size_t)DM * N_QKV * 2);       // 6 MB
  __bf16* Wp_s = (__bf16*)carve((size_t)DM * DM * 2);          // 2 MB
  __bf16* Qs   = (__bf16*)carve((size_t)MTOT * DM * 2);        // 8 MB
  __bf16* Kb   = (__bf16*)carve((size_t)MTOT * DM * 2);        // 8 MB
  __bf16* VT   = (__bf16*)carve((size_t)MTOT * DM * 2);        // 8 MB
  __bf16* AO   = (__bf16*)carve((size_t)MTOT * DM * 2);        // 8 MB
  unsigned* mb = (unsigned*)carve((size_t)TSEQ * MW * 4);      // 0.5 MB

  // stage 0: conversions / reorganizations (one-time, memory-light)
  {
    int n4 = (MTOT * DM) / 4;
    cvt_bf16_kernel<<<(n4 + 255) / 256, 256, 0, stream>>>(x, Xb, n4);
  }
  {
    int tq = (DM / KTILE) * (N_QKV / 16) * 32;  // 196608
    swz_w_kernel<<<(tq + 255) / 256, 256, 0, stream>>>(Wqkv, Wq_s, N_QKV);
    int tp = (DM / KTILE) * (DM / 16) * 32;     // 65536
    swz_w_kernel<<<(tp + 255) / 256, 256, 0, stream>>>(Wproj, Wp_s, DM);
  }
  {
    int tm = TSEQ * MW;  // 131072
    pack_mask_kernel<<<(tm + 255) / 256, 256, 0, stream>>>(mask, mb);
  }
  // stage 1: qkv projection (writes Q scaled by 1/sqrt(dh), K, V transposed)
  qkv_gemm_kernel<<<dim3(N_QKV / 256, MTOT / 64), 256, 0, stream>>>(
      Xb, Wq_s, bqkv, Qs, Kb, VT);
  // stage 2: flash attention, 8 cooperating waves per block (128 queries)
  attn_kernel<<<dim3(TSEQ / 128, NB * NHEAD), 256, 0, stream>>>(Qs, Kb, VT, mb, AO);
  // stage 3: output projection (f32 result)
  proj_gemm_kernel<<<dim3(DM / 256, MTOT / 64), 256, 0, stream>>>(
      AO, Wp_s, bproj, out);
}
